// MultiHeadSelfAttention_66760971649354
// MI455X (gfx1250) — matrix operand order repaired and checked by reading, not compiled, not run on hardware
//
#include <hip/hip_runtime.h>

typedef __attribute__((ext_vector_type(16))) __bf16 v16bf;
typedef __attribute__((ext_vector_type(8)))  float  v8f;
typedef __attribute__((ext_vector_type(2)))  __bf16 bf16x2;
typedef __attribute__((ext_vector_type(4)))  int    i32x4;

#define NU 512
#define HH 8
#define DK 64
#define BB 4
#define TT 1024
#define NTOK (BB*TT)   // 4096

#if __has_builtin(__builtin_amdgcn_global_load_async_to_lds_b128)
#define HAVE_ASYNC 1
#else
#define HAVE_ASYNC 0
#endif

// 16-byte global -> LDS copy; async (ASYNCcnt) when available.
// Builtin signature (from compiler diagnostic): (v4i as1*, v4i as3*, imm int, imm int)
__device__ __forceinline__ void async_cp16(void* lds, const void* g) {
#if HAVE_ASYNC
  __builtin_amdgcn_global_load_async_to_lds_b128(
      (__attribute__((address_space(1))) i32x4*)g,
      (__attribute__((address_space(3))) i32x4*)lds, 0, 0);
#else
  *reinterpret_cast<i32x4*>(lds) = *reinterpret_cast<const i32x4*>(g);
#endif
}

__device__ __forceinline__ void wait_async0() {
#if HAVE_ASYNC
#if __has_builtin(__builtin_amdgcn_s_wait_asynccnt)
  __builtin_amdgcn_s_wait_asynccnt(0);
#else
  asm volatile("s_wait_asynccnt 0x0" ::: "memory");
#endif
#endif
}

union FragBF { v16bf v; bf16x2 p[8]; };

__device__ __forceinline__ v8f wmma_bf16(v16bf a, v16bf b, v8f c) {
  return __builtin_amdgcn_wmma_f32_16x16x32_bf16(false, a, false, b, (short)0, c, false, false);
}

// A fragment: 16x32 bf16 tile, row-major, ld = row stride.
// lane L: row M = L&15; VGPR i holds K = (L>>4)*16 + 2i, 2i+1
__device__ __forceinline__ v16bf frag_a(const __bf16* t, int rowBase, int ld, int kBase, int lane) {
  FragBF f;
  const int half = lane >> 4, mr = lane & 15;
  const __bf16* rp = t + (size_t)(rowBase + mr) * ld + kBase;
#pragma unroll
  for (int i = 0; i < 8; ++i) {
    int k = (half << 4) + (i << 1);
    f.p[i] = *reinterpret_cast<const bf16x2*>(rp + k);
  }
  return f.v;
}

// B fragment: 32x16 bf16 (KxN); source stored tile[n*ld + k] (K contiguous per n).
// lane L: col N = L&15; VGPR i holds K = (L>>4)*16 + 2i, 2i+1
__device__ __forceinline__ v16bf frag_b(const __bf16* t, int nBase, int ld, int kBase, int lane) {
  FragBF f;
  const int half = lane >> 4, nc = lane & 15;
  const __bf16* rp = t + (size_t)(nBase + nc) * ld + kBase + (half << 4);
#pragma unroll
  for (int i = 0; i < 8; ++i)
    f.p[i] = *reinterpret_cast<const bf16x2*>(rp + (i << 1));
  return f.v;
}

__global__ void cvt_bf16(const float* __restrict__ s, __bf16* __restrict__ d, int n) {
  int i = blockIdx.x * blockDim.x + threadIdx.x;
  if (i < n) d[i] = (__bf16)s[i];
}

// ---------------- fused QKV projection: [4096,512] x 3x[512,512] -------------
__global__ void __launch_bounds__(256) qkv_gemm(
    const __bf16* __restrict__ xb,
    const __bf16* __restrict__ wq, const __bf16* __restrict__ wk, const __bf16* __restrict__ wv,
    const float* __restrict__ bq, const float* __restrict__ bk, const float* __restrict__ bv,
    __bf16* __restrict__ Qm, __bf16* __restrict__ Km, __bf16* __restrict__ Vm) {
  __shared__ __bf16 sX[2][128*32];
  __shared__ __bf16 sWq[2][64*32], sWk[2][64*32], sWv[2][64*32];
  const int tid = threadIdx.x, lane = tid & 31, wave = tid >> 5;
  const int m0 = blockIdx.x * 128, n0 = blockIdx.y * 64;
  v8f aq[4] = {}, ak[4] = {}, av[4] = {};

  auto stage = [&](int buf, int kk) {
#pragma unroll
    for (int i = tid; i < 512; i += 256) {           // X tile 128x32: async b128
      int r = i >> 2, c = (i & 3) << 3;
      async_cp16(&sX[buf][r*32 + c], &xb[(size_t)(m0 + r)*NU + kk + c]);
    }
    for (int i = tid; i < 2048; i += 256) {          // W tiles 32x64 -> [n][k]
      int k = i >> 6, n = i & 63;
      size_t g = (size_t)(kk + k)*NU + n0 + n;
      sWq[buf][n*32 + k] = wq[g];
      sWk[buf][n*32 + k] = wk[g];
      sWv[buf][n*32 + k] = wv[g];
    }
  };

  stage(0, 0);
  wait_async0();
  __syncthreads();
  for (int kk = 0; kk < NU; kk += 32) {
    int cur = (kk >> 5) & 1;
    if (kk + 32 < NU) stage(cur ^ 1, kk + 32);       // prefetch next tile
    v16bf a = frag_a(sX[cur], wave*16, 32, 0, lane);
#pragma unroll
    for (int nt = 0; nt < 4; ++nt) {
      aq[nt] = wmma_bf16(a, frag_b(sWq[cur], nt*16, 32, 0, lane), aq[nt]);
      ak[nt] = wmma_bf16(a, frag_b(sWk[cur], nt*16, 32, 0, lane), ak[nt]);
      av[nt] = wmma_bf16(a, frag_b(sWv[cur], nt*16, 32, 0, lane), av[nt]);
    }
    wait_async0();
    __syncthreads();
  }
  const int half = lane >> 4;
#pragma unroll
  for (int nt = 0; nt < 4; ++nt) {
    int col = n0 + nt*16 + (lane & 15);
#pragma unroll
    for (int r = 0; r < 8; ++r) {
      int row = m0 + wave*16 + r + half*8;
      size_t idx = (size_t)row*NU + col;
      Qm[idx] = (__bf16)(aq[nt][r] + bq[col]);
      Km[idx] = (__bf16)(ak[nt][r] + bk[col]);
      Vm[idx] = (__bf16)(av[nt][r] + bv[col]);
    }
  }
}

// ------------- relative-position scores: per q-position [32,64]x[64,T] ------
// grid: (1024 q positions, 4 key chunks of 256); single pass over fp32 pos_k.
__global__ void __launch_bounds__(256) pos_scores(
    const __bf16* __restrict__ Qm, const float* __restrict__ posk, __bf16* __restrict__ spos) {
  __shared__ __bf16 sA[32*64];
  const int tid = threadIdx.x, lane = tid & 31, wave = tid >> 5;
  const int q = blockIdx.x;
  const int k0 = blockIdx.y * 256 + wave * 32;
  {                                                  // A = Q rows for all (b,h)
    int bh = tid >> 3, c = (tid & 7) << 3;
    async_cp16(&sA[bh*64 + c],
               &Qm[(size_t)((bh >> 3)*TT + q)*NU + (bh & 7)*DK + c]);
  }
  wait_async0();
  __syncthreads();
  const int half = lane >> 4;
  v8f acc[2][2] = {};
#pragma unroll
  for (int ks = 0; ks < 2; ++ks) {
    v16bf a0 = frag_a(sA, 0,  64, ks*32, lane);
    v16bf a1 = frag_a(sA, 16, 64, ks*32, lane);
#pragma unroll
    for (int nt = 0; nt < 2; ++nt) {
      int key = k0 + nt*16 + (lane & 15);
      const float* rp = posk + ((size_t)q*TT + key)*DK + ks*32 + half*16;
      FragBF f;                                      // cvt fp32 -> bf16 on the fly
#pragma unroll
      for (int i = 0; i < 8; ++i) {
        bf16x2 t;
        t.x = (__bf16)rp[2*i];
        t.y = (__bf16)rp[2*i + 1];
        f.p[i] = t;
      }
      acc[0][nt] = wmma_bf16(a0, f.v, acc[0][nt]);
      acc[1][nt] = wmma_bf16(a1, f.v, acc[1][nt]);
    }
  }
#pragma unroll
  for (int mt = 0; mt < 2; ++mt)
#pragma unroll
    for (int nt = 0; nt < 2; ++nt)
#pragma unroll
      for (int r = 0; r < 8; ++r) {
        int bh  = mt*16 + r + half*8;
        int key = k0 + nt*16 + (lane & 15);
        spos[((size_t)bh*TT + q)*TT + key] = (__bf16)acc[mt][nt][r];
      }
}

// ---------------- flash attention with additive pos-score ---------------------
// grid: (T/128 q-blocks, B*H). Each wave owns a 16-query M-tile.
__global__ void __launch_bounds__(256) attn(
    const __bf16* __restrict__ Qm, const __bf16* __restrict__ Km, const __bf16* __restrict__ Vm,
    const __bf16* __restrict__ spos, __bf16* __restrict__ ctx) {
  __shared__ __bf16 sQ[128*64];
  __shared__ __bf16 sK[2][64*64];   // [key][d] : K^T B-frags (K-dim = d contiguous)
  __shared__ __bf16 sVt[2][64*64];  // [d][key] : V  B-frags (K-dim = key contiguous)
  __shared__ __bf16 sP[8][16*64];   // per-wave P relayout C-layout -> A-layout
  const int tid = threadIdx.x, lane = tid & 31, wave = tid >> 5;
  const int half = lane >> 4, nc = lane & 15;
  const int bh = blockIdx.y, b = bh >> 3, h = bh & 7;
  const int q0 = blockIdx.x * 128;

#pragma unroll
  for (int i = tid; i < 1024; i += 256) {            // Q tile 128x64: async b128
    int r = i >> 3, c = (i & 7) << 3;
    async_cp16(&sQ[r*64 + c], &Qm[(size_t)(b*TT + q0 + r)*NU + h*DK + c]);
  }
  auto stageKV = [&](int buf, int kb0) {
#pragma unroll
    for (int i = tid; i < 512; i += 256) {           // K tile 64x64: async b128
      int r = i >> 3, c = (i & 7) << 3;
      async_cp16(&sK[buf][r*64 + c], &Km[(size_t)(b*TT + kb0 + r)*NU + h*DK + c]);
    }
    for (int i = tid; i < 4096; i += 256) {          // transpose V while staging
      int n = i >> 6, d = i & 63;
      sVt[buf][d*64 + n] = Vm[(size_t)(b*TT + kb0 + n)*NU + h*DK + d];
    }
  };
  stageKV(0, 0);
  wait_async0();
  __syncthreads();

  v16bf aQ0 = frag_a(sQ, wave*16, 64, 0,  lane);
  v16bf aQ1 = frag_a(sQ, wave*16, 64, 32, lane);
  v8f accO[4] = {};
  float mrow[8], lrow[8];
#pragma unroll
  for (int r = 0; r < 8; ++r) { mrow[r] = -1e30f; lrow[r] = 0.f; }

  for (int kb0 = 0; kb0 < TT; kb0 += 64) {
    int cur = (kb0 >> 6) & 1;
    if (kb0 + 64 < TT) stageKV(cur ^ 1, kb0 + 64);   // prefetch next key block

    float st[4][8];
#pragma unroll
    for (int nt = 0; nt < 4; ++nt) {
      v8f s = {};
      s = wmma_bf16(aQ0, frag_b(sK[cur], nt*16, 64, 0,  lane), s);
      s = wmma_bf16(aQ1, frag_b(sK[cur], nt*16, 64, 32, lane), s);
      int key = kb0 + nt*16 + nc;
#pragma unroll
      for (int r = 0; r < 8; ++r) {
        int qq = q0 + wave*16 + r + half*8;
        float sp = (float)spos[((size_t)bh*TT + qq)*TT + key];
        st[nt][r] = 0.125f * (s[r] + sp);            // 1/sqrt(64)
      }
    }
    // online softmax: rows live in 16-lane groups (xor 1,2,4,8 stays in-group)
#pragma unroll
    for (int r = 0; r < 8; ++r) {
      float mx = st[0][r];
#pragma unroll
      for (int nt = 1; nt < 4; ++nt) mx = fmaxf(mx, st[nt][r]);
#pragma unroll
      for (int off = 1; off < 16; off <<= 1) mx = fmaxf(mx, __shfl_xor(mx, off, 32));
      float mnew  = fmaxf(mrow[r], mx);
      float alpha = __expf(mrow[r] - mnew);
      mrow[r] = mnew;
      float rs = 0.f;
#pragma unroll
      for (int nt = 0; nt < 4; ++nt) {
        float p = __expf(st[nt][r] - mnew);
        st[nt][r] = p;
        rs += p;
      }
#pragma unroll
      for (int off = 1; off < 16; off <<= 1) rs += __shfl_xor(rs, off, 32);
      lrow[r] = lrow[r]*alpha + rs;
#pragma unroll
      for (int dt = 0; dt < 4; ++dt) accO[dt][r] *= alpha;
    }
    // P: C-layout regs -> per-wave LDS -> A-layout frags (same-wave LDS is in-order)
#pragma unroll
    for (int nt = 0; nt < 4; ++nt)
#pragma unroll
      for (int r = 0; r < 8; ++r)
        sP[wave][(r + half*8)*64 + nt*16 + nc] = (__bf16)st[nt][r];
    v16bf aP0 = frag_a(sP[wave], 0, 64, 0,  lane);
    v16bf aP1 = frag_a(sP[wave], 0, 64, 32, lane);
#pragma unroll
    for (int dt = 0; dt < 4; ++dt) {
      accO[dt] = wmma_bf16(aP0, frag_b(sVt[cur], dt*16, 64, 0,  lane), accO[dt]);
      accO[dt] = wmma_bf16(aP1, frag_b(sVt[cur], dt*16, 64, 32, lane), accO[dt]);
    }
    wait_async0();
    __syncthreads();
  }
#pragma unroll
  for (int dt = 0; dt < 4; ++dt)
#pragma unroll
    for (int r = 0; r < 8; ++r) {
      int qq = q0 + wave*16 + r + half*8;
      int d  = dt*16 + nc;
      ctx[(size_t)(b*TT + qq)*NU + h*DK + d] = (__bf16)(accO[dt][r] / lrow[r]);
    }
}

// ---------------- output projection: ctx @ Wo + bo -> fp32 out ---------------
__global__ void __launch_bounds__(256) out_gemm(
    const __bf16* __restrict__ xb, const __bf16* __restrict__ w,
    const float* __restrict__ bias, float* __restrict__ out) {
  __shared__ __bf16 sX[2][128*32];
  __shared__ __bf16 sW[2][64*32];
  const int tid = threadIdx.x, lane = tid & 31, wave = tid >> 5;
  const int m0 = blockIdx.x * 128, n0 = blockIdx.y * 64;
  v8f acc[4] = {};

  auto stage = [&](int buf, int kk) {
#pragma unroll
    for (int i = tid; i < 512; i += 256) {
      int r = i >> 2, c = (i & 3) << 3;
      async_cp16(&sX[buf][r*32 + c], &xb[(size_t)(m0 + r)*NU + kk + c]);
    }
    for (int i = tid; i < 2048; i += 256) {
      int k = i >> 6, n = i & 63;
      sW[buf][n*32 + k] = w[(size_t)(kk + k)*NU + n0 + n];
    }
  };

  stage(0, 0);
  wait_async0();
  __syncthreads();
  for (int kk = 0; kk < NU; kk += 32) {
    int cur = (kk >> 5) & 1;
    if (kk + 32 < NU) stage(cur ^ 1, kk + 32);
    v16bf a = frag_a(sX[cur], wave*16, 32, 0, lane);
#pragma unroll
    for (int nt = 0; nt < 4; ++nt)
      acc[nt] = wmma_bf16(a, frag_b(sW[cur], nt*16, 32, 0, lane), acc[nt]);
    wait_async0();
    __syncthreads();
  }
  const int half = lane >> 4;
#pragma unroll
  for (int nt = 0; nt < 4; ++nt) {
    int col = n0 + nt*16 + (lane & 15);
#pragma unroll
    for (int r = 0; r < 8; ++r) {
      int row = m0 + wave*16 + r + half*8;
      out[(size_t)row*NU + col] = acc[nt][r] + bias[col];
    }
  }
}

extern "C" void kernel_launch(void* const* d_in, const int* in_sizes, int n_in,
                              void* d_out, int out_size, void* d_ws, size_t ws_size,
                              hipStream_t stream) {
  (void)in_sizes; (void)n_in; (void)out_size; (void)ws_size;
  const float* x    = (const float*)d_in[0];
  const float* posk = (const float*)d_in[1];
  const float* Wq   = (const float*)d_in[3];
  const float* bq   = (const float*)d_in[4];
  const float* Wk   = (const float*)d_in[5];
  const float* bk   = (const float*)d_in[6];
  const float* Wv   = (const float*)d_in[7];
  const float* bv   = (const float*)d_in[8];
  const float* Wo   = (const float*)d_in[9];
  const float* bo   = (const float*)d_in[10];
  float* out = (float*)d_out;
  char* ws = (char*)d_ws;

  __bf16* xb   = (__bf16*)(ws + 0);          // 4096*512*2        = 4 MiB
  __bf16* wqb  = (__bf16*)(ws + 4194304);    // 512*512*2         = 512 KiB
  __bf16* wkb  = (__bf16*)(ws + 4718592);
  __bf16* wvb  = (__bf16*)(ws + 5242880);
  __bf16* wob  = (__bf16*)(ws + 5767168);
  __bf16* Qm   = (__bf16*)(ws + 6291456);    // 4 MiB each
  __bf16* Km   = (__bf16*)(ws + 10485760);
  __bf16* Vm   = (__bf16*)(ws + 14680064);
  __bf16* spos = (__bf16*)(ws + 18874368);   // 32*1024*1024*2    = 64 MiB (L2-resident)
  __bf16* ctx  = (__bf16*)(ws + 85983232);   // 4 MiB; total ~86 MiB

  cvt_bf16<<<dim3((NTOK*NU + 255)/256), 256, 0, stream>>>(x,  xb,  NTOK*NU);
  cvt_bf16<<<dim3((NU*NU   + 255)/256), 256, 0, stream>>>(Wq, wqb, NU*NU);
  cvt_bf16<<<dim3((NU*NU   + 255)/256), 256, 0, stream>>>(Wk, wkb, NU*NU);
  cvt_bf16<<<dim3((NU*NU   + 255)/256), 256, 0, stream>>>(Wv, wvb, NU*NU);
  cvt_bf16<<<dim3((NU*NU   + 255)/256), 256, 0, stream>>>(Wo, wob, NU*NU);

  qkv_gemm  <<<dim3(NTOK/128, NU/64), 256, 0, stream>>>(xb, wqb, wkb, wvb, bq, bk, bv, Qm, Km, Vm);
  pos_scores<<<dim3(TT, 4),           256, 0, stream>>>(Qm, posk, spos);
  attn      <<<dim3(TT/128, BB*HH),   256, 0, stream>>>(Qm, Km, Vm, spos, ctx);
  out_gemm  <<<dim3(NTOK/128, NU/64), 256, 0, stream>>>(ctx, wob, bo, out);
}
